// GaussianDistribution2d_39960375722028
// MI455X (gfx1250) — matrix operand order repaired and checked by reading, not compiled, not run on hardware
//
#include <hip/hip_runtime.h>
#include <hip/hip_bf16.h>
#include <math.h>

// ---------------- constants ----------------
#define BSZ     32
#define IMG     224
#define PATCH   16
#define DM      48
#define HEADS   12
#define HDIM    4
#define MLPD    192
#define LAYERS  6
#define NPTS    48
#define NPATCH  196          // (224/16)^2
#define NTOK    244          // NPATCH + NPTS
#define HALF    24           // NPTS/2
#define PIX     (IMG*IMG)    // 50176

typedef __attribute__((ext_vector_type(16))) _Float16 v16h;
typedef __attribute__((ext_vector_type(8)))  float    v8f;

__device__ __forceinline__ float gelu_f(float x) {
    // exact GELU: 0.5*x*(1+erf(x/sqrt(2)))
    return 0.5f * x * (1.0f + erff(x * 0.70710678118654752f));
}

// =====================================================================
// WMMA GEMM:  C[M,N] = act( A[M,K] @ W[K,N] + bias )   (optionally += C)
// N, K, ACT, ACCUM, HASBIAS compile-time. One wave per 16x16 C tile.
// A-fragment: per lane two contiguous 8-float runs -> 4x b128 loads.
// B-fragment: 16 lane-coalesced strided loads, no predication.
// Epilogue fully branch-free; residual path clauses 8 loads -> 8 stores.
// =====================================================================
#define ACT_NONE  0
#define ACT_GELU  1
#define ACT_CLIP4 2

template<int N, int K, int ACT, bool ACCUM, bool HASBIAS>
__global__ void __launch_bounds__(256)
wmma_gemm_f16(const float* __restrict__ A, const float* __restrict__ W,
              const float* __restrict__ bias, float* __restrict__ C,
              int M, int rowsPerB, long batchStrideA, int lda)
{
    const int wave = threadIdx.x >> 5;
    const int lane = threadIdx.x & 31;
    const int tilesM = M >> 4;
    constexpr int tilesN = N / 16;
    const int tid = blockIdx.x * 8 + wave;
    if (tid >= tilesM * tilesN) return;          // wave-uniform exit
    const int tm = tid % tilesM;
    const int tn = tid / tilesM;

    const int  rloc  = lane & 15;
    const bool hi    = lane >= 16;
    const int  kbase = hi ? 8 : 0;

    // A row handled by this lane (lanes 16-31 reuse rows 0-15, high K-half)
    const int arow = tm * 16 + rloc;
    const long aoff = (long)(arow / rowsPerB) * batchStrideA
                    + (long)(arow % rowsPerB) * lda;
    const float* __restrict__ Arow = A + aoff;
    const int bcol = tn * 16 + rloc;

    v8f acc = {};
    #pragma unroll 2
    for (int k0 = 0; k0 < K; k0 += 32) {
        constexpr bool hasTail = (K % 32) != 0;
        const bool full = !hasTail || (k0 + 16) < K;   // constant-folds

        // ---- A fragment: lanes 0-15 K={k0..+7, k0+16..+23}
        //                  lanes16-31 K={k0+8..+15, k0+24..+31}
        const float4* a0p = reinterpret_cast<const float4*>(Arow + k0 + kbase);
        float4 x0 = a0p[0];
        float4 x1 = a0p[1];
        float4 x2 = make_float4(0.f, 0.f, 0.f, 0.f);
        float4 x3 = make_float4(0.f, 0.f, 0.f, 0.f);
        if (full) {
            const float4* a1p = reinterpret_cast<const float4*>(Arow + k0 + 16 + kbase);
            x2 = a1p[0];
            x3 = a1p[1];
        }
        v16h a;
        a[0]=(_Float16)x0.x;  a[1]=(_Float16)x0.y;  a[2]=(_Float16)x0.z;  a[3]=(_Float16)x0.w;
        a[4]=(_Float16)x1.x;  a[5]=(_Float16)x1.y;  a[6]=(_Float16)x1.z;  a[7]=(_Float16)x1.w;
        a[8]=(_Float16)x2.x;  a[9]=(_Float16)x2.y;  a[10]=(_Float16)x2.z; a[11]=(_Float16)x2.w;
        a[12]=(_Float16)x3.x; a[13]=(_Float16)x3.y; a[14]=(_Float16)x3.z; a[15]=(_Float16)x3.w;

        // ---- B fragment: same K order as A (lanes 0-15 K={k0..+7, k0+16..+23}, lanes16-31 +8)
        v16h b;
        if (full) {
            const float* Wp = W + (long)(k0 + kbase) * N + bcol;
            #pragma unroll
            for (int e = 0; e < 16; ++e) b[e] = (_Float16)Wp[(long)(e < 8 ? e : e + 8) * N];
        } else {
            // last 16-wide chunk: elements 8-15 (K>=k0+16) out of range -> 0
            const float* Wp = W + (long)(k0 + kbase) * N + bcol;
            #pragma unroll
            for (int e = 0; e < 16; ++e) b[e] = (e < 8) ? (_Float16)Wp[(long)e * N] : (_Float16)0.0f;
        }

        acc = __builtin_amdgcn_wmma_f32_16x16x32_f16(
                  false, a, false, b, (short)0, acc, false, false);
    }

    // C/D layout: VGPR r -> lanes 0-15: M=r, lanes 16-31: M=8+r; N = lane&15
    const int ocol  = tn * 16 + rloc;
    const int rbase = tm * 16 + (hi ? 8 : 0);
    const float bv  = HASBIAS ? bias[ocol] : 0.0f;

    float cold[8];
    if (ACCUM) {
        #pragma unroll
        for (int r = 0; r < 8; ++r)
            cold[r] = C[(long)(rbase + r) * N + ocol];    // 8 coalesced loads, 1 wait
    }
    #pragma unroll
    for (int r = 0; r < 8; ++r) {
        float v = acc[r] + bv;
        if (ACT == ACT_GELU)       v = gelu_f(v);
        else if (ACT == ACT_CLIP4) v = fminf(fmaxf(v, 0.0f), 4.0f);
        if (ACCUM) v += cold[r];
        C[(long)(rbase + r) * N + ocol] = v;
    }
}

// =====================================================================
// Patch conv: feat[b, p, d] = sum_{256} mask_patch * conv_w[d] + conv_b[d]
// =====================================================================
__global__ void __launch_bounds__(64)
conv_patch(const float* __restrict__ mask, const float* __restrict__ cw,
           const float* __restrict__ cb, float* __restrict__ feat)
{
    int bp = blockIdx.x;
    int b = bp / NPATCH, p = bp % NPATCH;
    int pr = p / 14, pc = p % 14;
    __shared__ float patch[256];
    const float* mb = mask + ((long)b * IMG + pr * PATCH) * IMG + pc * PATCH;
    for (int i = threadIdx.x; i < 256; i += 64) {
        int rr = i >> 4, cc = i & 15;
        patch[i] = mb[rr * IMG + cc];
    }
    __syncthreads();
    int d = threadIdx.x;
    if (d < DM) {
        float s = cb[d];
        const float* w = cw + d * 256;
        #pragma unroll 8
        for (int i = 0; i < 256; ++i) s += patch[i] * w[i];
        feat[((long)b * NTOK + p) * DM + d] = s;
    }
}

// points_embed -> feat rows [196, 244)
__global__ void points_embed(const float* __restrict__ pts,
                             const float* __restrict__ pw,
                             const float* __restrict__ pb,
                             float* __restrict__ feat)
{
    int idx = blockIdx.x * 256 + threadIdx.x;           // B*48*48
    if (idx >= BSZ * NPTS * DM) return;
    int d = idx % DM;
    int n = (idx / DM) % NPTS;
    int b = idx / (NPTS * DM);
    const float* pp = pts + ((long)b * NPTS + n) * 3;
    float s = pb[d] + pp[0]*pw[0*DM+d] + pp[1]*pw[1*DM+d] + pp[2]*pw[2*DM+d];
    feat[((long)b * NTOK + NPATCH + n) * DM + d] = s;
}

// =====================================================================
// LayerNorm over 48-dim rows; one wave per row (lane covers e, e+32)
// =====================================================================
__global__ void __launch_bounds__(256)
ln48(const float* __restrict__ x, float* __restrict__ out,
     const float* __restrict__ s, const float* __restrict__ bb, int rows)
{
    int wave = threadIdx.x >> 5, lane = threadIdx.x & 31;
    int row = blockIdx.x * 8 + wave;
    if (row >= rows) return;
    const float* xr = x + (long)row * DM;
    float v0 = xr[lane];
    float v1 = (lane < 16) ? xr[32 + lane] : 0.0f;
    float sum = v0 + v1;
    #pragma unroll
    for (int o = 16; o; o >>= 1) sum += __shfl_xor(sum, o, 32);
    float mean = sum * (1.0f / 48.0f);
    float d0 = v0 - mean;
    float d1 = (lane < 16) ? (v1 - mean) : 0.0f;
    float vs = d0 * d0 + d1 * d1;
    #pragma unroll
    for (int o = 16; o; o >>= 1) vs += __shfl_xor(vs, o, 32);
    float inv = rsqrtf(vs * (1.0f / 48.0f) + 1e-5f);
    float* orr = out + (long)row * DM;
    orr[lane] = d0 * inv * s[lane] + bb[lane];
    if (lane < 16) orr[32 + lane] = d1 * inv * s[32 + lane] + bb[32 + lane];
}

// row-wise L2 normalize, 48-dim rows, in place
__global__ void __launch_bounds__(256)
l2norm48(float* __restrict__ x, int rows)
{
    int wave = threadIdx.x >> 5, lane = threadIdx.x & 31;
    int row = blockIdx.x * 8 + wave;
    if (row >= rows) return;
    float* xr = x + (long)row * DM;
    float v0 = xr[lane];
    float v1 = (lane < 16) ? xr[32 + lane] : 0.0f;
    float ss = v0 * v0 + v1 * v1;
    #pragma unroll
    for (int o = 16; o; o >>= 1) ss += __shfl_xor(ss, o, 32);
    float inv = rsqrtf(ss);
    xr[lane] = v0 * inv;
    if (lane < 16) xr[32 + lane] = v1 * inv;
}

// =====================================================================
// Fused attention, head-dim 4, online softmax. qkv: (B,244,144) packed
// =====================================================================
__global__ void __launch_bounds__(256)
attn_fused(const float* __restrict__ qkv, float* __restrict__ y)
{
    int bh = blockIdx.x;
    int b = bh / HEADS, h = bh % HEADS;
    __shared__ float Ks[NTOK * HDIM];
    __shared__ float Vs[NTOK * HDIM];
    for (int i = threadIdx.x; i < NTOK * HDIM; i += 256) {
        int t = i >> 2, e = i & 3;
        long base = ((long)b * NTOK + t) * (3 * DM) + h * HDIM + e;
        Ks[i] = qkv[base + DM];
        Vs[i] = qkv[base + 2 * DM];
    }
    __syncthreads();
    int t = threadIdx.x;
    if (t < NTOK) {
        long qb = ((long)b * NTOK + t) * (3 * DM) + h * HDIM;
        float q0 = qkv[qb], q1 = qkv[qb+1], q2 = qkv[qb+2], q3 = qkv[qb+3];
        float mx = -1e30f, sum = 0.0f;
        float a0 = 0.f, a1 = 0.f, a2 = 0.f, a3 = 0.f;
        for (int m = 0; m < NTOK; ++m) {
            const float* kk = Ks + m * HDIM;
            float sc = 0.5f * (q0*kk[0] + q1*kk[1] + q2*kk[2] + q3*kk[3]); // scale=hd^-0.5
            float nmx  = fmaxf(mx, sc);
            float corr = expf(mx - nmx);
            float p    = expf(sc - nmx);
            const float* vv = Vs + m * HDIM;
            sum = sum * corr + p;
            a0 = a0 * corr + p * vv[0];
            a1 = a1 * corr + p * vv[1];
            a2 = a2 * corr + p * vv[2];
            a3 = a3 * corr + p * vv[3];
            mx = nmx;
        }
        float inv = 1.0f / sum;
        long yb = ((long)b * NTOK + t) * DM + h * HDIM;
        y[yb]   = a0 * inv; y[yb+1] = a1 * inv;
        y[yb+2] = a2 * inv; y[yb+3] = a3 * inv;
    }
}

// point_mask[b,n,p] = dot(mf[b,n,:], pf[b,p,:]); grid B*196, 64 threads
__global__ void __launch_bounds__(64)
pm_dots(const float* __restrict__ mf, const float* __restrict__ pf,
        float* __restrict__ pm)
{
    int bn = blockIdx.x;
    int b = bn / NPATCH, n = bn % NPATCH;
    __shared__ float mrow[DM];
    if (threadIdx.x < DM) mrow[threadIdx.x] = mf[((long)b*NPATCH + n)*DM + threadIdx.x];
    __syncthreads();
    int p = threadIdx.x;
    if (p < NPTS) {
        const float* pr = pf + ((long)b * NPTS + p) * DM;
        float s = 0.0f;
        #pragma unroll 8
        for (int d = 0; d < DM; ++d) s += mrow[d] * pr[d];
        pm[((long)b * NPATCH + n) * NPTS + p] = s;
    }
}

__global__ void gelu_ew(const float* __restrict__ x, float* __restrict__ o, int n)
{
    int i = blockIdx.x * 256 + threadIdx.x;
    if (i < n) o[i] = gelu_f(x[i]);
}

// =====================================================================
// Gaussian phase prep (tiny, 1 thread per (b, phase))
// =====================================================================
__global__ void phase_prep(const float* __restrict__ points,
                           const float* __restrict__ vars,
                           float* __restrict__ cx, float* __restrict__ cy,
                           float* __restrict__ vp, float* __restrict__ validf,
                           float* __restrict__ nvb)
{
    int bp = blockIdx.x;               // b*2 + phase
    if (threadIdx.x != 0) return;
    int b = bp >> 1, ph = bp & 1;
    int nv = 0;
    for (int j = 0; j < HALF; ++j) {
        const float* pt = points + ((long)b * NPTS + ph * HALF + j) * 3;
        float c0 = pt[0], c1 = pt[1];
        bool valid = fmaxf(c0, c1) > 0.0f;
        if (valid) nv++;
        int rank = nv - 1;
        if (rank < 0) rank = 0;
        if (rank > HALF - 1) rank = HALF - 1;
        int o = bp * HALF + j;
        cx[o] = c0;
        cy[o] = c1;
        vp[o] = vars[b * NPTS + ph * HALF + rank] + 4.0f;   // VAR_BIAS
        validf[o] = valid ? 1.0f : 0.0f;
    }
    nvb[bp] = (float)nv;
}

// out[bp, r, c] = sum_j valid * exp(-(dr^2+dc^2)/(2v^2))/(pi*2v^2) / max(nv,1)
__global__ void __launch_bounds__(256)
gauss_map(const float* __restrict__ cx, const float* __restrict__ cy,
          const float* __restrict__ vp, const float* __restrict__ validf,
          const float* __restrict__ nvb, float* __restrict__ out)
{
    int bp  = blockIdx.x;
    int idx = blockIdx.y * 256 + threadIdx.x;   // 196 * 256 == 50176 exactly
    float fr = (float)(idx / IMG);
    float fc = (float)(idx % IMG);
    float s = 0.0f;
    for (int j = 0; j < HALF; ++j) {
        int o = bp * HALF + j;
        if (validf[o] > 0.0f) {
            float v2 = 2.0f * vp[o] * vp[o];
            float dr = fr - cx[o], dc = fc - cy[o];
            s += expf(-(dr * dr + dc * dc) / v2) / (3.14159265358979f * v2);
        }
    }
    float nv = fmaxf(nvb[bp], 1.0f);
    out[(long)bp * PIX + idx] = s / nv;
}

// Per-image min/max normalize to [-1,1]; zero if no valid points.
__global__ void __launch_bounds__(256)
gauss_norm(float* __restrict__ out, const float* __restrict__ nvb)
{
    int bp = blockIdx.x;
    float* img = out + (long)bp * PIX;
    __shared__ float smn[256], smx[256];
    float mn = 1e30f, mx = -1e30f;
    for (int i = threadIdx.x; i < PIX; i += 256) {
        float v = img[i];
        mn = fminf(mn, v); mx = fmaxf(mx, v);
    }
    smn[threadIdx.x] = mn; smx[threadIdx.x] = mx;
    __syncthreads();
    for (int o = 128; o; o >>= 1) {
        if (threadIdx.x < o) {
            smn[threadIdx.x] = fminf(smn[threadIdx.x], smn[threadIdx.x + o]);
            smx[threadIdx.x] = fmaxf(smx[threadIdx.x], smx[threadIdx.x + o]);
        }
        __syncthreads();
    }
    mn = smn[0]; mx = smx[0];
    float denom = (mx > mn) ? (mx - mn) : 1.0f;
    bool zero = !(nvb[bp] > 0.0f);
    for (int i = threadIdx.x; i < PIX; i += 256) {
        float v = 2.0f * (img[i] - mn) / denom - 1.0f;
        img[i] = zero ? 0.0f : v;
    }
}

// =====================================================================
// host orchestration
// =====================================================================
static inline int gemm_grid(int M, int N) {
    int tiles = (M / 16) * (N / 16);
    return (tiles + 7) / 8;
}

extern "C" void kernel_launch(void* const* d_in, const int* in_sizes, int n_in,
                              void* d_out, int out_size, void* d_ws, size_t ws_size,
                              hipStream_t stream) {
    const float* mask    = (const float*)d_in[1];
    const float* points  = (const float*)d_in[2];
    const float* conv_w  = (const float*)d_in[3];
    const float* conv_b  = (const float*)d_in[4];
    const float* plin_w  = (const float*)d_in[5];
    const float* plin_b  = (const float*)d_in[6];
    const float* n1_s    = (const float*)d_in[7];
    const float* n1_b    = (const float*)d_in[8];
    const float* qkv_w   = (const float*)d_in[9];
    const float* qkv_b   = (const float*)d_in[10];
    const float* ap_w    = (const float*)d_in[11];
    const float* ap_b    = (const float*)d_in[12];
    const float* n2_s    = (const float*)d_in[13];
    const float* n2_b    = (const float*)d_in[14];
    const float* fc1_w   = (const float*)d_in[15];
    const float* fc1_b   = (const float*)d_in[16];
    const float* fc2_w   = (const float*)d_in[17];
    const float* fc2_b   = (const float*)d_in[18];
    const float* proj_m  = (const float*)d_in[19];
    const float* proj_p  = (const float*)d_in[20];
    const float* dn_s    = (const float*)d_in[21];
    const float* dn_b    = (const float*)d_in[22];
    const float* mn_s    = (const float*)d_in[23];
    const float* mn_b    = (const float*)d_in[24];
    const float* mlp_w   = (const float*)d_in[25];
    const float* mlp_b   = (const float*)d_in[26];

    float* ws = (float*)d_ws;
    long off = 0;
    float* feat = ws + off; off += (long)BSZ * NTOK * DM;          // 374784
    float* hbuf = ws + off; off += (long)BSZ * NTOK * DM;          // 374784
    float* mid  = ws + off; off += (long)BSZ * NTOK * MLPD;        // 1499136 (qkv 144 / fc1 192)
    float* ybuf = ws + off; off += (long)BSZ * NTOK * DM;          // 374784
    float* mf   = ws + off; off += (long)BSZ * NPATCH * DM;        // 301056
    float* pf   = ws + off; off += (long)BSZ * NPTS * DM;          // 73728
    float* pm   = ws + off; off += (long)BSZ * NPATCH * NPTS;      // 301056
    float* gbuf = ws + off; off += (long)BSZ * NPATCH * NPTS;      // 301056
    float* vars = ws + off; off += (long)BSZ * NPTS;               // 1536
    float* cxb  = ws + off; off += BSZ * 2 * HALF;
    float* cyb  = ws + off; off += BSZ * 2 * HALF;
    float* vpb  = ws + off; off += BSZ * 2 * HALF;
    float* vfb  = ws + off; off += BSZ * 2 * HALF;
    float* nvb  = ws + off; off += BSZ * 2;

    const int M   = BSZ * NTOK;        // 7808 (488 tiles)
    const int Mmf = BSZ * NPATCH;      // 6272
    const int Mpf = BSZ * NPTS;        // 1536
    const long fstride = (long)NTOK * DM;

    // embeddings
    conv_patch<<<BSZ * NPATCH, 64, 0, stream>>>(mask, conv_w, conv_b, feat);
    points_embed<<<(BSZ * NPTS * DM + 255) / 256, 256, 0, stream>>>(points, plin_w, plin_b, feat);

    // transformer
    for (int i = 0; i < LAYERS; ++i) {
        ln48<<<(M + 7) / 8, 256, 0, stream>>>(feat, hbuf, n1_s + i*DM, n1_b + i*DM, M);
        wmma_gemm_f16<144, 48, ACT_NONE, false, true>
            <<<gemm_grid(M, 144), 256, 0, stream>>>(
            hbuf, qkv_w + (long)i*DM*3*DM, qkv_b + i*3*DM, mid, M, M, 0, DM);
        attn_fused<<<BSZ * HEADS, 256, 0, stream>>>(mid, ybuf);
        wmma_gemm_f16<48, 48, ACT_NONE, true, true>
            <<<gemm_grid(M, 48), 256, 0, stream>>>(
            ybuf, ap_w + (long)i*DM*DM, ap_b + i*DM, feat, M, M, 0, DM);     // residual add
        ln48<<<(M + 7) / 8, 256, 0, stream>>>(feat, hbuf, n2_s + i*DM, n2_b + i*DM, M);
        wmma_gemm_f16<192, 48, ACT_GELU, false, true>
            <<<gemm_grid(M, 192), 256, 0, stream>>>(
            hbuf, fc1_w + (long)i*DM*MLPD, fc1_b + i*MLPD, mid, M, M, 0, DM);
        wmma_gemm_f16<48, 192, ACT_NONE, true, true>
            <<<gemm_grid(M, 48), 256, 0, stream>>>(
            mid, fc2_w + (long)i*MLPD*DM, fc2_b + i*DM, feat, M, M, 0, MLPD); // residual add
    }

    // head
    ln48<<<(M + 7) / 8, 256, 0, stream>>>(feat, hbuf, dn_s, dn_b, M);
    wmma_gemm_f16<48, 48, ACT_NONE, false, false>
        <<<gemm_grid(Mmf, 48), 256, 0, stream>>>(
        hbuf, proj_m, nullptr, mf, Mmf, NPATCH, fstride, DM);
    wmma_gemm_f16<48, 48, ACT_NONE, false, false>
        <<<gemm_grid(Mpf, 48), 256, 0, stream>>>(
        hbuf + (long)NPATCH*DM, proj_p, nullptr, pf, Mpf, NPTS, fstride, DM);
    l2norm48<<<(Mmf + 7) / 8, 256, 0, stream>>>(mf, Mmf);
    l2norm48<<<(Mpf + 7) / 8, 256, 0, stream>>>(pf, Mpf);
    pm_dots<<<BSZ * NPATCH, 64, 0, stream>>>(mf, pf, pm);
    ln48<<<(Mmf + 7) / 8, 256, 0, stream>>>(pm, pm, mn_s, mn_b, Mmf);   // in-place safe (wave-local)
    gelu_ew<<<(BSZ*NPATCH*NPTS + 255) / 256, 256, 0, stream>>>(pm, gbuf, BSZ*NPATCH*NPTS);
    wmma_gemm_f16<48, NPATCH*NPTS, ACT_CLIP4, false, true>
        <<<gemm_grid(BSZ, 48), 256, 0, stream>>>(
        gbuf, mlp_w, mlp_b, vars, BSZ, BSZ, 0, NPATCH*NPTS);

    // gaussian phases -> d_out (B,2,224,224)
    phase_prep<<<BSZ * 2, 32, 0, stream>>>(points, vars, cxb, cyb, vpb, vfb, nvb);
    gauss_map<<<dim3(BSZ * 2, PIX / 256), 256, 0, stream>>>(cxb, cyb, vpb, vfb, nvb, (float*)d_out);
    gauss_norm<<<BSZ * 2, 256, 0, stream>>>((float*)d_out, nvb);

    (void)in_sizes; (void)n_in; (void)out_size; (void)ws_size;
}
